// QuantumModel_33200097198091
// MI455X (gfx1250) — matrix operand order repaired and checked by reading, not compiled, not run on hardware
//
#include <hip/hip_runtime.h>
#include <hip/hip_bf16.h>
#include <math.h>

typedef __attribute__((ext_vector_type(16))) __bf16 v16bf;
typedef __attribute__((ext_vector_type(8)))  float  v8f;

// d_ws layout:
//   float  wsf[0..511]   : U as interleaved (re,im), row-major 16x16
//   float  wsf[512..543] : cw[b][t] = sum_q sign(b,q)*post_w[q][t]
//   byte 4096.. : bf16 W1 fragments, 50 frags (25 ksteps x 2 ngroups) * 512 ushort
#define WS_CW_FOFF   512
#define WS_FRAG_BYTE 4096

__device__ inline float2 cmulc(float2 a, float2 b) {
  return make_float2(a.x*b.x - a.y*b.y, a.x*b.y + a.y*b.x);
}

// ---------------- setup: build U, cw, and pre-swizzled bf16 W1 fragments ----
__global__ void qm_setup(const float* __restrict__ w1,
                         const float* __restrict__ sw,   // (2,4,3)
                         const float* __restrict__ tw,   // (1,4,3)
                         const float* __restrict__ pw,   // (4,2)
                         float* __restrict__ wsf,
                         unsigned short* __restrict__ wsfrag)
{
  __shared__ float2 U[16][16];
  __shared__ float2 T[16][16];
  if (threadIdx.x == 0) {
    for (int i = 0; i < 16; ++i)
      for (int j = 0; j < 16; ++j)
        U[i][j] = make_float2(i == j ? 1.f : 0.f, 0.f);

    for (int layer = 0; layer < 3; ++layer) {
      const float* w; int r;
      if (layer < 2) { w = sw + layer*12; r = layer % 3 + 1; }
      else           { w = tw;            r = 1; }
      // single-qubit rotations, left-multiplied
      for (int i = 0; i < 4; ++i) {
        float phi = w[i*3+0], th = w[i*3+1], om = w[i*3+2];
        float cc = cosf(0.5f*th), ssn = sinf(0.5f*th);
        float a0 = -0.5f*(phi+om), a1 = 0.5f*(phi-om);
        float2 m00 = make_float2( cosf(a0)*cc,   sinf(a0)*cc);
        float2 m01 = make_float2(-cosf(a1)*ssn, -sinf(a1)*ssn);
        float2 m10 = make_float2( cosf(a1)*ssn, -sinf(a1)*ssn);
        float2 m11 = make_float2( cosf(a0)*cc,  -sinf(a0)*cc);
        int bit = 1 << (3 - i);
        for (int i0 = 0; i0 < 16; ++i0) {
          if (i0 & bit) continue;
          int i1 = i0 | bit;
          for (int j = 0; j < 16; ++j) {
            float2 u0 = U[i0][j], u1 = U[i1][j];
            float2 p0 = cmulc(m00, u0), p1 = cmulc(m01, u1);
            float2 p2 = cmulc(m10, u0), p3 = cmulc(m11, u1);
            U[i0][j] = make_float2(p0.x + p1.x, p0.y + p1.y);
            U[i1][j] = make_float2(p2.x + p3.x, p2.y + p3.y);
          }
        }
      }
      // CNOT ring, left-multiplied (row permutation)
      for (int i = 0; i < 4; ++i) {
        int c = i, t = (i + r) & 3;
        for (int b = 0; b < 16; ++b)
          for (int j = 0; j < 16; ++j) T[b][j] = U[b][j];
        for (int b = 0; b < 16; ++b) {
          int b2 = ((b >> (3 - c)) & 1) ? (b ^ (1 << (3 - t))) : b;
          for (int j = 0; j < 16; ++j) U[b2][j] = T[b][j];
        }
      }
    }
    for (int b = 0; b < 16; ++b)
      for (int k = 0; k < 16; ++k) {
        wsf[(b*16+k)*2+0] = U[b][k].x;
        wsf[(b*16+k)*2+1] = U[b][k].y;
      }
    for (int b = 0; b < 16; ++b)
      for (int t = 0; t < 2; ++t) {
        float acc = 0.f;
        for (int q = 0; q < 4; ++q) {
          float sgn = ((b >> (3 - q)) & 1) ? -1.f : 1.f;
          acc += sgn * pw[q*2+t];
        }
        wsf[WS_CW_FOFF + b*2 + t] = acc;
      }
  }

  // bf16 B-fragments: lanes 0-15 hold K=ks*32+{0-7,16-23}, lanes 16-31 K=ks*32+{8-15,24-31};
  // lane's column N = ng*16 + (lane&15); K padded to 800 with zeros.
  for (int idx = threadIdx.x; idx < 50*512; idx += blockDim.x) {
    int f = idx >> 9, rem = idx & 511;
    int lane = rem >> 4, i = rem & 15;
    int ks = f >> 1, ng = f & 1;
    int K = ks*32 + ((i >> 3) << 4) + ((lane >> 4) << 3) + (i & 7);
    int N = ng*16 + (lane & 15);
    float v = (K < 784) ? w1[K*32 + N] : 0.f;
    union { float fv; unsigned u; } uu; uu.fv = v;
    unsigned rb = uu.u + 0x7FFFu + ((uu.u >> 16) & 1u);   // RNE f32->bf16
    wsfrag[idx] = (unsigned short)(rb >> 16);
  }
}

// ---------------- main fused kernel -----------------------------------------
__global__ __launch_bounds__(256)
void qm_main(const float* __restrict__ x,
             const float* __restrict__ b1,
             const float* __restrict__ lng,
             const float* __restrict__ lnb,
             const float* __restrict__ w2,
             const float* __restrict__ b2,
             const float* __restrict__ pb,
             const float* __restrict__ wsf,
             const unsigned short* __restrict__ wsfrag,
             float2* __restrict__ out,
             int B)
{
  __shared__ float sU[512];
  __shared__ float scw[32];
  __shared__ float sw2[128];
  __shared__ float sb2[4];
  __shared__ float spb[2];
  __shared__ float sh[128*33];   // h_norm, stride 33 to avoid bank conflicts

  int tid = threadIdx.x;
  for (int i = tid; i < 512; i += 256) sU[i] = wsf[i];
  if (tid < 32)  scw[tid] = wsf[WS_CW_FOFF + tid];
  if (tid < 128) sw2[tid] = w2[tid];
  if (tid < 4)   sb2[tid] = b2[tid];
  if (tid < 2)   spb[tid] = pb[tid];

  int lane = tid & 31, wid = tid >> 5;
  int m = lane & 15, half = lane >> 4;
  int rowg = blockIdx.x*128 + wid*16 + m;
  int rowc = (rowg < B) ? rowg : 0;
  const float* xrow = x + (size_t)rowc * 784;

  // GEMM: h(16x32) = x(16x784) @ W1(784x32), bf16 WMMA, f32 accumulate
  v8f c0 = {0,0,0,0,0,0,0,0};
  v8f c1 = {0,0,0,0,0,0,0,0};
  for (int ks = 0; ks < 25; ++ks) {
    int kb = ks*32 + half*8;      // A layout: half0 -> K {0-7,16-23}, half1 -> {8-15,24-31}
    float4 f0 = *(const float4*)(xrow + kb);
    float4 f1 = *(const float4*)(xrow + kb + 4);
    float4 f2, f3;
    if (ks < 24) {
      f2 = *(const float4*)(xrow + kb + 16);
      f3 = *(const float4*)(xrow + kb + 20);
      __builtin_prefetch(xrow + kb + 288, 0, 1);
    } else {                       // K = 784..799 zero pad
      f2 = make_float4(0.f, 0.f, 0.f, 0.f);
      f3 = make_float4(0.f, 0.f, 0.f, 0.f);
    }
    v16bf a;
    a[0]=(__bf16)f0.x;  a[1]=(__bf16)f0.y;  a[2]=(__bf16)f0.z;  a[3]=(__bf16)f0.w;
    a[4]=(__bf16)f1.x;  a[5]=(__bf16)f1.y;  a[6]=(__bf16)f1.z;  a[7]=(__bf16)f1.w;
    a[8]=(__bf16)f2.x;  a[9]=(__bf16)f2.y;  a[10]=(__bf16)f2.z; a[11]=(__bf16)f2.w;
    a[12]=(__bf16)f3.x; a[13]=(__bf16)f3.y; a[14]=(__bf16)f3.z; a[15]=(__bf16)f3.w;

    v16bf bb0 = *(const v16bf*)(wsfrag + (ks*2+0)*512 + lane*16);
    v16bf bb1 = *(const v16bf*)(wsfrag + (ks*2+1)*512 + lane*16);

    c0 = __builtin_amdgcn_wmma_f32_16x16x32_bf16(false, a, false, bb0, (short)0, c0, false, false);
    c1 = __builtin_amdgcn_wmma_f32_16x16x32_bf16(false, a, false, bb1, (short)0, c1, false, false);
  }

  // bias + relu + LayerNorm (row reductions = 16-lane butterflies, wave32)
  int n0 = m;
  float g0 = lng[n0], g1 = lng[n0+16];
  float e0 = lnb[n0], e1 = lnb[n0+16];
  float bi0 = b1[n0], bi1 = b1[n0+16];
  #pragma unroll
  for (int j = 0; j < 8; ++j) {
    float v0 = fmaxf(c0[j] + bi0, 0.f);
    float v1 = fmaxf(c1[j] + bi1, 0.f);
    float s  = v0 + v1;
    float q2 = v0*v0 + v1*v1;
    #pragma unroll
    for (int d = 1; d < 16; d <<= 1) {
      s  += __shfl_xor(s,  d, 32);
      q2 += __shfl_xor(q2, d, 32);
    }
    float mu  = s * 0.03125f;
    float var = q2 * 0.03125f - mu*mu;
    float inv = rsqrtf(var + 1e-5f);
    int r = wid*16 + half*8 + j;   // C layout: half0 -> rows 0-7, half1 -> rows 8-15
    sh[r*33 + n0]      = (v0 - mu)*inv*g0 + e0;
    sh[r*33 + n0 + 16] = (v1 - mu)*inv*g1 + e1;
  }
  __syncthreads();

  // per-row tail (both halves duplicate; half 0 stores)
  int rl = wid*16 + m;
  float hr[32];
  #pragma unroll
  for (int n = 0; n < 32; ++n) hr[n] = sh[rl*33 + n];

  float zz[4];
  #pragma unroll
  for (int q = 0; q < 4; ++q) {
    float acc = sb2[q];
    #pragma unroll
    for (int n = 0; n < 32; ++n) acc = fmaf(hr[n], sw2[n*4 + q], acc);
    zz[q] = tanhf(acc);
  }
  float cs[4], sn[4];
  #pragma unroll
  for (int q = 0; q < 4; ++q) {
    float ang = zz[q] * 1.5707963267948966f;
    cs[q] = cosf(ang); sn[q] = sinf(ang);
  }
  float psi[16];
  #pragma unroll
  for (int b = 0; b < 16; ++b) {
    float p = ((b>>3)&1 ? sn[0] : cs[0]);
    p *= ((b>>2)&1 ? sn[1] : cs[1]);
    p *= ((b>>1)&1 ? sn[2] : cs[2]);
    p *= ((b>>0)&1 ? sn[3] : cs[3]);
    psi[b] = p;
  }
  float o0 = spb[0], o1 = spb[1];
  #pragma unroll 4
  for (int b = 0; b < 16; ++b) {
    float re = 0.f, im = 0.f;
    #pragma unroll
    for (int k = 0; k < 16; ++k) {
      float ur = sU[(b*16+k)*2+0];
      float ui = sU[(b*16+k)*2+1];
      re = fmaf(ur, psi[k], re);
      im = fmaf(ui, psi[k], im);
    }
    float p = re*re + im*im;
    o0 = fmaf(p, scw[b*2+0], o0);
    o1 = fmaf(p, scw[b*2+1], o1);
  }
  if (half == 0 && rowg < B) out[rowg] = make_float2(o0, o1);
}

// ---------------- launch ----------------------------------------------------
extern "C" void kernel_launch(void* const* d_in, const int* in_sizes, int n_in,
                              void* d_out, int out_size, void* d_ws, size_t ws_size,
                              hipStream_t stream)
{
  const float* x   = (const float*)d_in[0];
  const float* w1  = (const float*)d_in[1];
  const float* b1  = (const float*)d_in[2];
  const float* lng = (const float*)d_in[3];
  const float* lnb = (const float*)d_in[4];
  const float* w2  = (const float*)d_in[5];
  const float* b2  = (const float*)d_in[6];
  const float* sw  = (const float*)d_in[7];
  const float* tw  = (const float*)d_in[8];
  const float* pw  = (const float*)d_in[9];
  const float* pb  = (const float*)d_in[10];

  int B = in_sizes[0] / 784;
  float* wsf = (float*)d_ws;
  unsigned short* wsfrag = (unsigned short*)((char*)d_ws + WS_FRAG_BYTE);

  qm_setup<<<1, 256, 0, stream>>>(w1, sw, tw, pw, wsf, wsfrag);
  int nblk = (B + 127) / 128;
  qm_main<<<nblk, 256, 0, stream>>>(x, b1, lng, lnb, w2, b2, pb,
                                    wsf, wsfrag, (float2*)d_out, B);
}
